// EnhancedLocalComplexAttention_21277267985089
// MI455X (gfx1250) — compile-verified, matrix operand order checked by reading, not run on hardware
//
#include <hip/hip_runtime.h>

// ---------------------------------------------------------------------------
// EnhancedLocalComplexAttention for MI455X (gfx1250, wave32, WMMA f16)
// Pipeline (all matrix math on v_wmma_f32_16x16x32_f16, f32 accumulate):
//   0) one-shot f32 -> f16 conversion of x and all weight planes
//   1) q/k/v = x @ W^T (complex GEMM, f16 in, f16 out)
//   2) windowed attention (f16 in, f32 softmax, f16 out)
//   3) out = attn @ wo^T (complex GEMM, f16 in, f32 out to d_out)
// GEMM tile staging uses GLOBAL_LOAD_ASYNC_TO_LDS_B128 (ASYNCcnt path) when
// the toolchain exposes the builtin, else plain b128 copies.
// ---------------------------------------------------------------------------

typedef __attribute__((ext_vector_type(16))) _Float16 v16h;
typedef __attribute__((ext_vector_type(8)))  _Float16 v8h;
typedef __attribute__((ext_vector_type(4)))  _Float16 v4h;
typedef __attribute__((ext_vector_type(8)))  float    v8f;
typedef __attribute__((ext_vector_type(4)))  int      v4i;

#define DIM    512
#define NTOK   8192          // b * n = 2 * 4096
#define WSZ    128           // window size
#define HEADS  8
#define DHEAD  64
#define SCALE  0.125f        // 64^-0.5

// ---- async global->LDS staging (CDNA5), guarded ---------------------------
#if __has_builtin(__builtin_amdgcn_global_load_async_to_lds_b128) && \
    __has_builtin(__builtin_amdgcn_s_wait_asynccnt)
#define HAVE_ASYNC_B128 1
#else
#define HAVE_ASYNC_B128 0
#endif

__device__ __forceinline__ void copy16_g2l(const _Float16* g, _Float16* l) {
#if HAVE_ASYNC_B128
  // builtin signature (from toolchain diagnostic): (int4 AS1*, int4 AS3*, imm, imm)
  __builtin_amdgcn_global_load_async_to_lds_b128(
      (__attribute__((address_space(1))) v4i*)g,
      (__attribute__((address_space(3))) v4i*)l,
      0, 0);
#else
  *(v8h*)l = *(const v8h*)g;
#endif
}

__device__ __forceinline__ void copy_fence() {
#if HAVE_ASYNC_B128
  __builtin_amdgcn_s_wait_asynccnt(0);
#endif
}

// ---- fragment helpers (16x16x32 f16 WMMA) ---------------------------------
// A-matrix 16x32 layout (ISA 7.12.2): lane<16 holds row (lane), K={0..7,16..23};
// lane>=16 holds row (lane-16), K={8..15,24..31}. Each fragment is two
// contiguous 8-half (16B) runs -> compiler emits 2x b128 loads.
__device__ __forceinline__ v16h frag_h(const _Float16* __restrict__ base, int ld,
                                       int lane) {
  const int r = lane & 15, g = lane >> 4;
  const _Float16* p = base + r * ld + g * 8;
  v16h f;
#pragma unroll
  for (int e = 0; e < 8; ++e) f[e] = p[e];
#pragma unroll
  for (int e = 0; e < 8; ++e) f[8 + e] = p[16 + e];
  return f;
}

__device__ __forceinline__ v16h frag_neg(v16h a) {
  v16h f;
#pragma unroll
  for (int e = 0; e < 16; ++e) f[e] = -a[e];  // packed v_xor 0x80008000
  return f;
}

__device__ __forceinline__ v8f wmma_f16(v16h a, v16h b, v8f c) {
  return __builtin_amdgcn_wmma_f32_16x16x32_f16(false, a, false, b, (short)0, c,
                                                false, false);
}

// ---------------------------------------------------------------------------
// Kernel 0: f32 -> f16 conversion (vectorized 4-wide)
// ---------------------------------------------------------------------------
__global__ __launch_bounds__(256)
void cvt_f32_f16(const float* __restrict__ src, _Float16* __restrict__ dst, int n4) {
  int i = blockIdx.x * 256 + threadIdx.x;
  if (i < n4) {
    float4 v = ((const float4*)src)[i];
    v4h h;
    h[0] = (_Float16)v.x; h[1] = (_Float16)v.y;
    h[2] = (_Float16)v.z; h[3] = (_Float16)v.w;
    ((v4h*)dst)[i] = h;
  }
}

// ---------------------------------------------------------------------------
// Complex GEMM: O = A @ W^T  (A:[M,K] cplx f16, W:[N,K] cplx f16, row-major)
//   O_re = Ar Wr^T - Ai Wi^T ; O_im = Ar Wi^T + Ai Wr^T
// Block tile 64x128, 8 waves, wave tile 32x32, K-step 32, LDS-staged.
// ---------------------------------------------------------------------------
#define BM  64
#define BN  128
#define BK  32
#define LDA 40   // LDS row stride in halves: 80B rows, 16B-aligned chunks

template <typename OutT>
__global__ __launch_bounds__(256)
void cplx_gemm_h(const _Float16* __restrict__ Ar, const _Float16* __restrict__ Ai,
                 const _Float16* __restrict__ Wr, const _Float16* __restrict__ Wi,
                 OutT* __restrict__ Or, OutT* __restrict__ Oi) {
  __shared__ _Float16 sAr[BM][LDA], sAi[BM][LDA];
  __shared__ _Float16 sBr[BN][LDA], sBi[BN][LDA];

  const int tid  = threadIdx.x;
  const int lane = tid & 31;
  const int wv   = tid >> 5;         // 0..7
  const int wm   = wv >> 2;          // 0..1 : wave row in block
  const int wn   = wv & 3;           // 0..3 : wave col in block
  const int m0   = blockIdx.y * BM;
  const int n0   = blockIdx.x * BN;

  // staging coordinates: 16B chunks of 8 halves
  const int ra  = tid >> 2, ca = (tid & 3) * 8;      // A: 64 rows x 4 chunks
  const int rb0 = tid >> 1, cb = (tid & 1) * 16;     // B: 128 rows x 2x16 halves

  v8f acc_re[2][2] = {}, acc_im[2][2] = {};

  for (int k0 = 0; k0 < DIM; k0 += BK) {
    __syncthreads();
    // stage A tile 64x32 halves: one 16B chunk per thread per component
    copy16_g2l(Ar + (size_t)(m0 + ra) * DIM + k0 + ca, &sAr[ra][ca]);
    copy16_g2l(Ai + (size_t)(m0 + ra) * DIM + k0 + ca, &sAi[ra][ca]);
    // stage W tile 128x32 halves: two 16B chunks per thread per component
    copy16_g2l(Wr + (size_t)(n0 + rb0) * DIM + k0 + cb,     &sBr[rb0][cb]);
    copy16_g2l(Wr + (size_t)(n0 + rb0) * DIM + k0 + cb + 8, &sBr[rb0][cb + 8]);
    copy16_g2l(Wi + (size_t)(n0 + rb0) * DIM + k0 + cb,     &sBi[rb0][cb]);
    copy16_g2l(Wi + (size_t)(n0 + rb0) * DIM + k0 + cb + 8, &sBi[rb0][cb + 8]);
    // prefetch next K tile into cache while WMMAs run (global_prefetch_b8)
    if (k0 + BK < DIM) {
      __builtin_prefetch((const void*)(Ar + (size_t)(m0 + ra) * DIM + k0 + BK), 0, 0);
      __builtin_prefetch((const void*)(Wr + (size_t)(n0 + rb0) * DIM + k0 + BK), 0, 0);
    }
    copy_fence();        // s_wait_asynccnt 0 (async path only)
    __syncthreads();

    v16h a_re[2], a_im[2], a_imn[2], b_re[2], b_im[2];
#pragma unroll
    for (int i = 0; i < 2; ++i) {
      a_re[i]  = frag_h(&sAr[wm * 32 + i * 16][0], LDA, lane);
      a_im[i]  = frag_h(&sAi[wm * 32 + i * 16][0], LDA, lane);
      a_imn[i] = frag_neg(a_im[i]);   // f16 WMMA has no A-negate modifier
    }
#pragma unroll
    for (int j = 0; j < 2; ++j) {
      b_re[j] = frag_h(&sBr[wn * 32 + j * 16][0], LDA, lane);
      b_im[j] = frag_h(&sBi[wn * 32 + j * 16][0], LDA, lane);
    }
#pragma unroll
    for (int i = 0; i < 2; ++i)
#pragma unroll
      for (int j = 0; j < 2; ++j) {
        acc_re[i][j] = wmma_f16(a_re[i],  b_re[j], acc_re[i][j]);
        acc_re[i][j] = wmma_f16(a_imn[i], b_im[j], acc_re[i][j]);
        acc_im[i][j] = wmma_f16(a_re[i],  b_im[j], acc_im[i][j]);
        acc_im[i][j] = wmma_f16(a_im[i],  b_re[j], acc_im[i][j]);
      }
  }

  // epilogue: C layout -> lane l, vgpr r : row = r + 8*(l>>4), col = l&15
  const int lg = lane >> 4, ln = lane & 15;
#pragma unroll
  for (int i = 0; i < 2; ++i)
#pragma unroll
    for (int j = 0; j < 2; ++j)
#pragma unroll
      for (int r = 0; r < 8; ++r) {
        size_t row = (size_t)(m0 + wm * 32 + i * 16 + r + 8 * lg);
        int    col = n0 + wn * 32 + j * 16 + ln;
        Or[row * DIM + col] = (OutT)acc_re[i][j][r];
        Oi[row * DIM + col] = (OutT)acc_im[i][j][r];
      }
}

// ---------------------------------------------------------------------------
// Windowed complex attention: one workgroup per (b, head, window).
//   S = (qr kr^T + qi ki^T) * scale + rel_bias    (128x128, f32 accum)
//   P = softmax_rows(S)  -> LDS f16
//   O_re = P v_re ; O_im = P v_im                 (V transposed via LDS)
// q/k fragments come straight from global f16 (2x b128 per fragment);
// V is transposed through LDS in two 32-wide d-halves (keeps LDS ~53KB).
// ---------------------------------------------------------------------------
#define LDP 136  // padded LDS stride in halves (272B, 16B-aligned)

__global__ __launch_bounds__(256)
void local_attn_h(const _Float16* __restrict__ Qr, const _Float16* __restrict__ Qi,
                  const _Float16* __restrict__ Kr, const _Float16* __restrict__ Ki,
                  const _Float16* __restrict__ Vr, const _Float16* __restrict__ Vi,
                  const float* __restrict__ bias,
                  _Float16* __restrict__ AOr, _Float16* __restrict__ AOi) {
  __shared__ _Float16 sP[WSZ][LDP];           // probabilities (f16)
  __shared__ _Float16 sVtr[32][LDP], sVti[32][LDP];  // V^T, one 32-wide d-half
  __shared__ float    sbias[2 * WSZ + 1];

  const int tid  = threadIdx.x;
  const int lane = tid & 31;
  const int wv   = tid >> 5;                 // 0..7 : 16-row strip
  const int blk  = blockIdx.x;               // b*h*nw = 512 blocks
  const int w    = blk & 31;                 // window
  const int h    = (blk >> 5) & 7;           // head
  const int bb   = blk >> 8;                 // batch

  if (tid < 2 * WSZ + 1) sbias[tid] = bias[tid];

  const size_t base = (size_t)(bb * 4096 + w * WSZ) * DIM + h * DHEAD;
  const _Float16* qr = Qr + base; const _Float16* qi = Qi + base;
  const _Float16* kr = Kr + base; const _Float16* ki = Ki + base;
  const _Float16* vr = Vr + base; const _Float16* vi = Vi + base;

  __syncthreads();

  // ---- S = real(q . conj(k)) : 8 n-tiles x (2 K-steps x 2 components) -----
  v8f accS[8] = {};
#pragma unroll
  for (int kc = 0; kc < 2; ++kc) {
    v16h aqr = frag_h(qr + (size_t)(wv * 16) * DIM + kc * 32, DIM, lane);
    v16h aqi = frag_h(qi + (size_t)(wv * 16) * DIM + kc * 32, DIM, lane);
#pragma unroll
    for (int t = 0; t < 8; ++t) {
      v16h bkr = frag_h(kr + (size_t)(t * 16) * DIM + kc * 32, DIM, lane);
      v16h bki = frag_h(ki + (size_t)(t * 16) * DIM + kc * 32, DIM, lane);
      accS[t] = wmma_f16(aqr, bkr, accS[t]);
      accS[t] = wmma_f16(aqi, bki, accS[t]);
    }
  }

  const int lg = lane >> 4, ln = lane & 15;

  // scale + relative position bias (|j-i| <= 127 inside a window: no clip)
#pragma unroll
  for (int t = 0; t < 8; ++t)
#pragma unroll
    for (int r = 0; r < 8; ++r) {
      int row = wv * 16 + r + 8 * lg;
      int col = t * 16 + ln;
      accS[t][r] = accS[t][r] * SCALE + sbias[col - row + WSZ];
    }

  // row softmax: each row lives in one 16-lane half-group, 8 values per lane
  float inv_sum[8];
#pragma unroll
  for (int r = 0; r < 8; ++r) {
    float m = -3.0e38f;
#pragma unroll
    for (int t = 0; t < 8; ++t) m = fmaxf(m, accS[t][r]);
#pragma unroll
    for (int s = 1; s <= 8; s <<= 1) m = fmaxf(m, __shfl_xor(m, s, 32));
    float sum = 0.f;
#pragma unroll
    for (int t = 0; t < 8; ++t) {
      float e = __expf(accS[t][r] - m);
      accS[t][r] = e;
      sum += e;
    }
#pragma unroll
    for (int s = 1; s <= 8; s <<= 1) sum += __shfl_xor(sum, s, 32);
    inv_sum[r] = 1.0f / sum;
  }

  // P -> LDS as f16 so it can be re-read in A-fragment layout
#pragma unroll
  for (int t = 0; t < 8; ++t)
#pragma unroll
    for (int r = 0; r < 8; ++r) {
      int row = wv * 16 + r + 8 * lg;
      int col = t * 16 + ln;
      sP[row][col] = (_Float16)(accS[t][r] * inv_sum[r]);
    }

  // ---- O = P @ V over two 32-wide d-halves of V^T ------------------------
  v8f accOr[4] = {}, accOi[4] = {};
#pragma unroll
  for (int half = 0; half < 2; ++half) {
    __syncthreads();  // (first pass: P visible; later: previous V^T reads done)
    // stage V^T half: 128 k-rows x 32 d -> sVt[d][k]; 2 chunks/thread/component
#pragma unroll
    for (int i = 0; i < 2; ++i) {
      int e  = tid + i * 256;         // 512 chunks of 8 halves
      int kk = e >> 2;                // 0..127
      int d0 = (e & 3) * 8;           // 0..24
      v8h hr = *(const v8h*)(vr + (size_t)kk * DIM + half * 32 + d0);
      v8h hi = *(const v8h*)(vi + (size_t)kk * DIM + half * 32 + d0);
#pragma unroll
      for (int j = 0; j < 8; ++j) {
        sVtr[d0 + j][kk] = hr[j];
        sVti[d0 + j][kk] = hi[j];
      }
    }
    __syncthreads();

#pragma unroll
    for (int kc = 0; kc < 4; ++kc) {
      v16h ap = frag_h(&sP[wv * 16][kc * 32], LDP, lane);
#pragma unroll
      for (int dt2 = 0; dt2 < 2; ++dt2) {
        int dt = half * 2 + dt2;
        v16h bvr = frag_h(&sVtr[dt2 * 16][kc * 32], LDP, lane);
        v16h bvi = frag_h(&sVti[dt2 * 16][kc * 32], LDP, lane);
        accOr[dt] = wmma_f16(ap, bvr, accOr[dt]);
        accOi[dt] = wmma_f16(ap, bvi, accOi[dt]);
      }
    }
  }

  // write head-merged attention output [b, n, 512] (f16)
#pragma unroll
  for (int dt = 0; dt < 4; ++dt)
#pragma unroll
    for (int r = 0; r < 8; ++r) {
      size_t row = base + (size_t)(wv * 16 + r + 8 * lg) * DIM;
      AOr[row + dt * 16 + ln] = (_Float16)accOr[dt][r];
      AOi[row + dt * 16 + ln] = (_Float16)accOi[dt][r];
    }
}

// ---------------------------------------------------------------------------
extern "C" void kernel_launch(void* const* d_in, const int* in_sizes, int n_in,
                              void* d_out, int out_size, void* d_ws, size_t ws_size,
                              hipStream_t stream) {
  (void)in_sizes; (void)n_in; (void)out_size; (void)ws_size;

  const float* rbias = (const float*)d_in[10];

  const size_t PLANE = (size_t)NTOK * DIM;   // 4,194,304 elements
  const size_t WMAT  = (size_t)DIM * DIM;    //   262,144 elements

  _Float16* hws = (_Float16*)d_ws;
  // f16 planes in workspace
  _Float16* xh[2];  xh[0] = hws;              xh[1] = hws + PLANE;
  _Float16* wh[8];                            // wq_re..wo_im
  for (int i = 0; i < 8; ++i) wh[i] = hws + 2 * PLANE + (size_t)i * WMAT;
  _Float16* qkvah = hws + 2 * PLANE + 8 * WMAT;
  _Float16* q_re = qkvah + 0 * PLANE; _Float16* q_im = qkvah + 1 * PLANE;
  _Float16* k_re = qkvah + 2 * PLANE; _Float16* k_im = qkvah + 3 * PLANE;
  _Float16* v_re = qkvah + 4 * PLANE; _Float16* v_im = qkvah + 5 * PLANE;
  _Float16* a_re = qkvah + 6 * PLANE; _Float16* a_im = qkvah + 7 * PLANE;

  // 0) convert x (inputs 0..1) and weights (inputs 2..9) to f16
  for (int i = 0; i < 2; ++i)
    cvt_f32_f16<<<(int)(PLANE / 4 / 256), 256, 0, stream>>>(
        (const float*)d_in[i], xh[i], (int)(PLANE / 4));
  for (int i = 0; i < 8; ++i)
    cvt_f32_f16<<<(int)(WMAT / 4 / 256), 256, 0, stream>>>(
        (const float*)d_in[2 + i], wh[i], (int)(WMAT / 4));

  dim3 ggrid(DIM / BN, NTOK / BM);           // (4, 128)

  // 1) projections (f16 out)
  cplx_gemm_h<_Float16><<<ggrid, 256, 0, stream>>>(xh[0], xh[1], wh[0], wh[1], q_re, q_im);
  cplx_gemm_h<_Float16><<<ggrid, 256, 0, stream>>>(xh[0], xh[1], wh[2], wh[3], k_re, k_im);
  cplx_gemm_h<_Float16><<<ggrid, 256, 0, stream>>>(xh[0], xh[1], wh[4], wh[5], v_re, v_im);

  // 2) windowed attention
  local_attn_h<<<dim3(2 * HEADS * (4096 / WSZ)), 256, 0, stream>>>(
      q_re, q_im, k_re, k_im, v_re, v_im, rbias, a_re, a_im);

  // 3) output projection (f32 out, straight into d_out planes)
  float* out_re = (float*)d_out;
  float* out_im = out_re + PLANE;
  cplx_gemm_h<float><<<ggrid, 256, 0, stream>>>(a_re, a_im, wh[6], wh[7], out_re, out_im);
}
